// LSTM_FIRST_44822278701287
// MI455X (gfx1250) — matrix operand order repaired and checked by reading, not compiled, not run on hardware
//
#include <hip/hip_runtime.h>
#include <hip/hip_bf16.h>
#include <math.h>

// ---------------- problem constants ----------------
#define T_STEPS 512
#define NB      64     // batch
#define NI      512    // input size
#define NH      512    // hidden size
#define NG      2048   // 4*H gates
#define NBLK    16     // persistent blocks (8 per direction, must be co-resident)
#define NWAVE   16     // waves per block (wave32) -> 512 threads

typedef __attribute__((ext_vector_type(8)))  float  v8f;
typedef __attribute__((ext_vector_type(8)))  __bf16 v8bf;
typedef __attribute__((ext_vector_type(16))) __bf16 v16bf;

__device__ __forceinline__ unsigned short f2bf(float f) {
  unsigned u = __float_as_uint(f);
  u += 0x7FFFu + ((u >> 16) & 1u);   // round-to-nearest-even
  return (unsigned short)(u >> 16);
}

__device__ __forceinline__ float sigf(float x) {
  return 1.0f / (1.0f + __expf(-x));
}

// Load a WMMA bf16 A-fragment (16x32 tile, this lane's 16 values) from a
// row-major bf16 row. Lane l<16 -> row m0+l, K offsets {0..15};
// lane l>=16 -> row m0+(l-16), K offsets {16..31}. Both halves are
// contiguous 16-byte runs in row-major storage.
__device__ __forceinline__ v16bf load_afrag(const unsigned short* rowbase, int koff) {
  v8bf lo = *(const v8bf*)(rowbase + koff);
  v8bf hi = *(const v8bf*)(rowbase + koff + 8);
  v16bf a;
#pragma unroll
  for (int i = 0; i < 8; ++i) { a[i] = lo[i]; a[i + 8] = hi[i]; }
  return a;
}

// ---------------- prep kernels ----------------
__global__ void zero_u32_kernel(unsigned* __restrict__ p, long n) {
  long i = (long)blockIdx.x * blockDim.x + threadIdx.x;
  if (i < n) p[i] = 0u;
}

// x (B,T,I) fp32 -> xb (T,B,I) bf16
__global__ void pack_x_kernel(const float* __restrict__ x, unsigned short* __restrict__ xb) {
  long idx = (long)blockIdx.x * blockDim.x + threadIdx.x;
  if (idx >= (long)T_STEPS * NB * NI) return;
  int  i = (int)(idx % NI);
  long r = idx / NI;
  int  b = (int)(r % NB);
  int  t = (int)(r / NB);
  xb[idx] = f2bf(x[((long)b * T_STEPS + t) * NI + i]);
}

__global__ void cvt_bf16_kernel(const float* __restrict__ s, unsigned short* __restrict__ d, long n) {
  long i = (long)blockIdx.x * blockDim.x + threadIdx.x;
  if (i < n) d[i] = f2bf(s[i]);
}

__global__ void bias_kernel(const float* __restrict__ bi1, const float* __restrict__ bh1,
                            const float* __restrict__ bi2, const float* __restrict__ bh2,
                            float* __restrict__ bias) {
  int idx = blockIdx.x * blockDim.x + threadIdx.x;
  if (idx >= 2 * NG) return;
  int d = idx >> 11, n = idx & (NG - 1);
  bias[idx] = d ? (bi2[n] + bh2[n]) : (bi1[n] + bh1[n]);
}

// ---------------- per-direction grid barrier (8 participating blocks) ----------------
__device__ __forceinline__ void gsync(unsigned* cnt, unsigned* gen, unsigned nb) {
  __threadfence();
  __syncthreads();
  if (threadIdx.x == 0) {
    unsigned g = __hip_atomic_load(gen, __ATOMIC_ACQUIRE, __HIP_MEMORY_SCOPE_AGENT);
    unsigned c = __hip_atomic_fetch_add(cnt, 1u, __ATOMIC_ACQ_REL, __HIP_MEMORY_SCOPE_AGENT);
    if (c == nb - 1u) {
      __hip_atomic_store(cnt, 0u, __ATOMIC_RELAXED, __HIP_MEMORY_SCOPE_AGENT);
      __hip_atomic_fetch_add(gen, 1u, __ATOMIC_ACQ_REL, __HIP_MEMORY_SCOPE_AGENT);
    } else {
      while (__hip_atomic_load(gen, __ATOMIC_ACQUIRE, __HIP_MEMORY_SCOPE_AGENT) == g)
        __builtin_amdgcn_s_sleep(1);
    }
  }
  __syncthreads();
}

// ---------------- persistent LSTM scan ----------------
// 16 blocks x 16 waves = 256 waves. Wave gw: dir = gw>>7 (blocks 0-7 fwd,
// 8-15 bwd; each direction syncs only among its own 8 blocks), batch-tile
// mt = (gw&127)>>5, hidden-tile jt = gw&31. Each wave owns a 16(batch)x16(hidden)
// patch; computes all 4 gate tiles with v_wmma_f32_16x16x32_bf16 (K=1024:
// x-projection + recurrence fused) and keeps the cell state c in VGPRs for
// all 512 steps.
__global__ void __launch_bounds__(NWAVE * 32, 1)
LSTM_scan_kernel(const unsigned short* __restrict__ xb,
                 const unsigned short* __restrict__ wih,
                 const unsigned short* __restrict__ whh,
                 const float* __restrict__ bias,
                 unsigned short* __restrict__ hbuf,
                 float* __restrict__ out,
                 unsigned* __restrict__ bar) {
  const int lane  = threadIdx.x & 31;
  const int wave  = threadIdx.x >> 5;
  const int gw    = blockIdx.x * NWAVE + wave;    // 0..255
  const int dir   = gw >> 7;                      // 0 fwd, 1 bwd (block-aligned)
  const int patch = gw & 127;
  const int m0    = (patch >> 5) * 16;            // batch row base
  const int j0    = (patch & 31) * 16;            // hidden col base
  const int lhalf = lane >> 4;
  const int l15   = lane & 15;

  unsigned* dbar = bar + dir * 32;                // separate barrier per direction

  // per-lane combined bias per gate (column j0 + l15)
  float bg[4];
#pragma unroll
  for (int g = 0; g < 4; ++g) bg[g] = bias[dir * NG + g * NH + j0 + l15];

  // B-fragment row pointers: column n; lane holds 16 contiguous K values
  const unsigned short* wihp[4];
  const unsigned short* whhp[4];
#pragma unroll
  for (int g = 0; g < 4; ++g) {
    size_t n = (size_t)dir * NG + (size_t)g * NH + j0 + l15;
    wihp[g] = wih + n * NI + 16 * lhalf;
    whhp[g] = whh + n * NH + 16 * lhalf;
  }

  unsigned short* hb0  = hbuf + (size_t)dir * 2 * NB * NH;
  const int arow  = m0 + l15;       // A-fragment row for this lane
  const int akoff = 16 * lhalf;     // A-fragment K base within 32-chunk

  v8f cacc;
#pragma unroll
  for (int e = 0; e < 8; ++e) cacc[e] = 0.0f;

  for (int s = 0; s < T_STEPS; ++s) {
    const int t = dir ? (T_STEPS - 1 - s) : s;
    const unsigned short* hread  = hb0 + (size_t)(s & 1)        * NB * NH;
    unsigned short*       hwrite = hb0 + (size_t)((s & 1) ^ 1)  * NB * NH;

    v8f a0, a1, a2, a3;
#pragma unroll
    for (int e = 0; e < 8; ++e) { a0[e] = bg[0]; a1[e] = bg[1]; a2[e] = bg[2]; a3[e] = bg[3]; }

    // x-projection part: K = NI
    const unsigned short* xrow = xb + ((size_t)t * NB + arow) * NI;
    {
      // prefetch next step's x row while we compute
      int tn = dir ? (t - 1) : (t + 1);
      if (tn >= 0 && tn < T_STEPS)
        __builtin_prefetch(xb + ((size_t)tn * NB + arow) * NI, 0, 1);
    }
#pragma unroll 4
    for (int kc = 0; kc < NI / 32; ++kc) {
      v16bf a  = load_afrag(xrow, kc * 32 + akoff);
      v16bf b0 = *(const v16bf*)(wihp[0] + kc * 32);
      v16bf b1 = *(const v16bf*)(wihp[1] + kc * 32);
      v16bf b2 = *(const v16bf*)(wihp[2] + kc * 32);
      v16bf b3 = *(const v16bf*)(wihp[3] + kc * 32);
      a0 = __builtin_amdgcn_wmma_f32_16x16x32_bf16(false, a, false, b0, (short)0, a0, false, false);
      a1 = __builtin_amdgcn_wmma_f32_16x16x32_bf16(false, a, false, b1, (short)0, a1, false, false);
      a2 = __builtin_amdgcn_wmma_f32_16x16x32_bf16(false, a, false, b2, (short)0, a2, false, false);
      a3 = __builtin_amdgcn_wmma_f32_16x16x32_bf16(false, a, false, b3, (short)0, a3, false, false);
    }

    // recurrent part: K = NH  (step 0 reads zero-initialized hbuf -> h0 = 0)
    const unsigned short* hrow = hread + (size_t)arow * NH;
#pragma unroll 4
    for (int kc = 0; kc < NH / 32; ++kc) {
      v16bf a  = load_afrag(hrow, kc * 32 + akoff);
      v16bf b0 = *(const v16bf*)(whhp[0] + kc * 32);
      v16bf b1 = *(const v16bf*)(whhp[1] + kc * 32);
      v16bf b2 = *(const v16bf*)(whhp[2] + kc * 32);
      v16bf b3 = *(const v16bf*)(whhp[3] + kc * 32);
      a0 = __builtin_amdgcn_wmma_f32_16x16x32_bf16(false, a, false, b0, (short)0, a0, false, false);
      a1 = __builtin_amdgcn_wmma_f32_16x16x32_bf16(false, a, false, b1, (short)0, a1, false, false);
      a2 = __builtin_amdgcn_wmma_f32_16x16x32_bf16(false, a, false, b2, (short)0, a2, false, false);
      a3 = __builtin_amdgcn_wmma_f32_16x16x32_bf16(false, a, false, b3, (short)0, a3, false, false);
    }

    // gate nonlinearities + cell update; C-layout: lanes<16 -> M=e, lanes>=16 -> M=8+e
#pragma unroll
    for (int e = 0; e < 8; ++e) {
      float ig = sigf(a0[e]);
      float fg = sigf(a1[e]);
      float gg = tanhf(a2[e]);
      float og = sigf(a3[e]);
      float c  = fg * cacc[e] + ig * gg;
      cacc[e]  = c;
      float h  = og * tanhf(c);
      int row  = m0 + e + 8 * lhalf;
      int col  = j0 + l15;
      atomicAdd(out + ((size_t)t * NB + row) * NH + col, 0.5f * h);
      hwrite[row * NH + col] = f2bf(h);
    }

    gsync(dbar, dbar + 1, (unsigned)(NBLK / 2));
  }
}

// ---------------- host launcher ----------------
extern "C" void kernel_launch(void* const* d_in, const int* in_sizes, int n_in,
                              void* d_out, int out_size, void* d_ws, size_t ws_size,
                              hipStream_t stream) {
  (void)in_sizes; (void)n_in; (void)ws_size;
  const float* x    = (const float*)d_in[0];
  const float* Wih1 = (const float*)d_in[1];
  const float* Whh1 = (const float*)d_in[2];
  const float* bih1 = (const float*)d_in[3];
  const float* bhh1 = (const float*)d_in[4];
  const float* Wih2 = (const float*)d_in[5];
  const float* Whh2 = (const float*)d_in[6];
  const float* bih2 = (const float*)d_in[7];
  const float* bhh2 = (const float*)d_in[8];
  float* out = (float*)d_out;

  // workspace layout (all 256-byte aligned), total ~40.3 MB
  const long XB_ELEMS = (long)T_STEPS * NB * NI;     // 16,777,216 bf16
  const long W_ELEMS  = (long)NG * NI;               // 1,048,576 bf16 per matrix
  char* ws = (char*)d_ws;
  unsigned*       bar  = (unsigned*)(ws + 0);                            // 256 B (2 barriers)
  unsigned short* xb   = (unsigned short*)(ws + 256);                    // 32 MB
  unsigned short* wih  = (unsigned short*)((char*)xb  + XB_ELEMS * 2);   // 4 MB (2 dirs)
  unsigned short* whh  = (unsigned short*)((char*)wih + 2 * W_ELEMS * 2);// 4 MB (2 dirs)
  float*          bias = (float*)((char*)whh + 2 * W_ELEMS * 2);         // 16 KB
  unsigned short* hbuf = (unsigned short*)((char*)bias + 2 * NG * 4);    // 256 KB

  const int TB = 256;
  // zero barriers, output (atomics accumulate), and h double-buffers (h0 = 0)
  zero_u32_kernel<<<1, 64, 0, stream>>>(bar, 64);
  zero_u32_kernel<<<(int)(((long)out_size + TB - 1) / TB), TB, 0, stream>>>((unsigned*)out, (long)out_size);
  {
    long hwords = (long)2 * 2 * NB * NH / 2;  // ushort count / 2 = dwords
    zero_u32_kernel<<<(int)((hwords + TB - 1) / TB), TB, 0, stream>>>((unsigned*)hbuf, hwords);
  }

  pack_x_kernel<<<(int)((XB_ELEMS + TB - 1) / TB), TB, 0, stream>>>(x, xb);
  int wblk = (int)((W_ELEMS + TB - 1) / TB);
  cvt_bf16_kernel<<<wblk, TB, 0, stream>>>(Wih1, wih,            W_ELEMS);
  cvt_bf16_kernel<<<wblk, TB, 0, stream>>>(Wih2, wih + W_ELEMS,  W_ELEMS);
  cvt_bf16_kernel<<<wblk, TB, 0, stream>>>(Whh1, whh,            W_ELEMS);
  cvt_bf16_kernel<<<wblk, TB, 0, stream>>>(Whh2, whh + W_ELEMS,  W_ELEMS);
  bias_kernel<<<(2 * NG + TB - 1) / TB, TB, 0, stream>>>(bih1, bhh1, bih2, bhh2, bias);

  LSTM_scan_kernel<<<NBLK, NWAVE * 32, 0, stream>>>(xb, wih, whh, bias, hbuf, out, bar);
}
